// Block1D_20444044329745
// MI455X (gfx1250) — matrix operand order repaired and checked by reading, not compiled, not run on hardware
//
#include <hip/hip_runtime.h>
#include <math.h>

// ---------------- problem constants ----------------
#define BB 8
#define NN 1024
#define DD 768
#define HH 12
#define DH 64
#define MLPH 3072
#define MROWS (BB * NN)   // 8192
#define SCALE 0.125f      // Dh^-0.5

typedef _Float16 h16;
typedef __attribute__((ext_vector_type(16))) _Float16 v16h;
typedef __attribute__((ext_vector_type(8)))  _Float16 v8h;
typedef __attribute__((ext_vector_type(8)))  float    v8f;

union AF { v16h v; v8h h[2]; };

#define WMMA_F16(a, b, c) \
  __builtin_amdgcn_wmma_f32_16x16x32_f16(false, (a), false, (b), (short)0, (c), false, false)

// A fragment: 16(M) x 32(K), row-major source, lda = ld (in halfs).
// lane l (0..31): m = l&15, hi = l>>4 ; holds K = {hi*8..hi*8+7, 16+hi*8..16+hi*8+7}
__device__ inline v16h load_frag_row(const h16* __restrict__ base, int ld, int k0, int lane) {
  const int m = lane & 15, hi = lane >> 4;
  const h16* p = base + (size_t)m * ld + k0 + hi * 8;
  AF f;
  f.h[0] = *(const v8h*)p;
  f.h[1] = *(const v8h*)(p + 16);
  return f.v;
}

// B fragment: 32(K) x 16(N) where B[k][n] = Wrow[n][k] (row-major W, ldw in halfs).
// lane l: n = l&15, hi = l>>4 ; holds K = {hi*8..hi*8+7, 16+hi*8..16+hi*8+7}
__device__ inline v16h load_frag_col(const h16* __restrict__ basecol, int ld, int k0, int lane) {
  const int n = lane & 15, hi = lane >> 4;
  const h16* p = basecol + (size_t)n * ld + k0 + hi * 8;
  AF f;
  f.h[0] = *(const v8h*)p;
  f.h[1] = *(const v8h*)(p + 16);
  return f.v;
}

// ---------------- fp32 -> f16 convert ----------------
__global__ void f32_to_f16_kernel(const float* __restrict__ src, h16* __restrict__ dst, int n) {
  int i = blockIdx.x * blockDim.x + threadIdx.x;
  int stride = gridDim.x * blockDim.x;
  for (; i < n; i += stride) dst[i] = (h16)src[i];
}

// ---------------- LayerNorm (row = one block), output f16 ----------------
__global__ __launch_bounds__(256) void ln_kernel(const float* __restrict__ x,
                                                 const float* __restrict__ gw,
                                                 const float* __restrict__ gb,
                                                 h16* __restrict__ out) {
  const size_t row = blockIdx.x;
  const float* xr = x + row * DD;
  float s = 0.f, ss = 0.f;
  for (int c = threadIdx.x; c < DD; c += 256) { float v = xr[c]; s += v; ss += v * v; }
  for (int off = 1; off < 32; off <<= 1) { s += __shfl_xor(s, off); ss += __shfl_xor(ss, off); }
  __shared__ float rs[8], rss[8];
  if ((threadIdx.x & 31) == 0) { rs[threadIdx.x >> 5] = s; rss[threadIdx.x >> 5] = ss; }
  __syncthreads();
  s = 0.f; ss = 0.f;
  for (int i = 0; i < 8; ++i) { s += rs[i]; ss += rss[i]; }
  const float mu = s * (1.f / DD);
  const float var = ss * (1.f / DD) - mu * mu;
  const float rstd = rsqrtf(var + 1e-5f);
  for (int c = threadIdx.x; c < DD; c += 256)
    out[row * DD + c] = (h16)((xr[c] - mu) * rstd * gw[c] + gb[c]);
}

// ---------------- generic WMMA GEMM: C[M,Nc] = A[M,K] @ W[Nc,K]^T ----------------
// block = 256 threads = 8 waves; block tile 128(M) x 64(N); wave tile 32x32.
// Explicit K-loop double buffering: fragment loads for k+32 are issued before the
// 4 WMMAs of step k, so s_wait_loadcnt is covered by matrix math.
// MODE 0: outF = acc + bias + (res?)        (fp32 out)
// MODE 2: QKV scatter -> q_h (SCALE folded), k_h, v_t(transposed)
// MODE 3: outH = gelu_exact(acc + bias)     (f16 out)
template <int MODE>
__global__ __launch_bounds__(256) void gemm_f16_kernel(
    const h16* __restrict__ A, const h16* __restrict__ W,
    int M, int Nc, int K,
    const float* __restrict__ bias, const float* __restrict__ res,
    float* __restrict__ outF, h16* __restrict__ outH,
    h16* __restrict__ qh, h16* __restrict__ kh, h16* __restrict__ vt) {
  const int lane = threadIdx.x & 31;
  const int wv = threadIdx.x >> 5;
  const int row0 = blockIdx.y * 128 + (wv >> 1) * 32;
  const int col0 = blockIdx.x * 64 + (wv & 1) * 32;

  v8f acc00 = {}, acc01 = {}, acc10 = {}, acc11 = {};
  const h16* Ar0 = A + (size_t)row0 * K;
  const h16* Ar1 = A + (size_t)(row0 + 16) * K;
  const h16* Wc0 = W + (size_t)col0 * K;
  const h16* Wc1 = W + (size_t)(col0 + 16) * K;

  // prologue: fragments for k0 = 0
  v16h a0 = load_frag_row(Ar0, K, 0, lane);
  v16h a1 = load_frag_row(Ar1, K, 0, lane);
  v16h b0 = load_frag_col(Wc0, K, 0, lane);
  v16h b1 = load_frag_col(Wc1, K, 0, lane);

  for (int k0 = 0; k0 < K - 32; k0 += 32) {
    const int kn = k0 + 32;
    // speculative prefetch two tiles ahead (silently dropped if OOB)
    __builtin_prefetch(Ar0 + (size_t)(lane & 15) * K + kn + 32, 0, 1);
    __builtin_prefetch(Wc0 + (size_t)(lane & 15) * K + kn + 32, 0, 1);
    // issue next-step loads before this step's math
    v16h na0 = load_frag_row(Ar0, K, kn, lane);
    v16h na1 = load_frag_row(Ar1, K, kn, lane);
    v16h nb0 = load_frag_col(Wc0, K, kn, lane);
    v16h nb1 = load_frag_col(Wc1, K, kn, lane);
    acc00 = WMMA_F16(a0, b0, acc00);
    acc01 = WMMA_F16(a0, b1, acc01);
    acc10 = WMMA_F16(a1, b0, acc10);
    acc11 = WMMA_F16(a1, b1, acc11);
    a0 = na0; a1 = na1; b0 = nb0; b1 = nb1;
  }
  // steady-state tail
  acc00 = WMMA_F16(a0, b0, acc00);
  acc01 = WMMA_F16(a0, b1, acc01);
  acc10 = WMMA_F16(a1, b0, acc10);
  acc11 = WMMA_F16(a1, b1, acc11);

  const int cn = lane & 15, hi = lane >> 4;
  auto epi = [&](const v8f& acc, int r0, int c0) {
    #pragma unroll
    for (int r = 0; r < 8; ++r) {
      const int grow = r0 + r + 8 * hi;
      const int gcol = c0 + cn;
      float val = acc[r];
      if (MODE == 0) {
        val += bias[gcol];
        if (res) val += res[(size_t)grow * Nc + gcol];
        outF[(size_t)grow * Nc + gcol] = val;
      } else if (MODE == 2) {
        const int b = grow >> 10, n = grow & 1023;
        if (gcol < DD) {
          const int h = gcol >> 6, d = gcol & 63;
          qh[(((size_t)(b * HH + h)) * NN + n) * DH + d] = (h16)(val * SCALE);
        } else if (gcol < 2 * DD) {
          const int c2 = gcol - DD, h = c2 >> 6, d = c2 & 63;
          kh[(((size_t)(b * HH + h)) * NN + n) * DH + d] = (h16)val;
        } else {
          const int c2 = gcol - 2 * DD, h = c2 >> 6, d = c2 & 63;
          vt[(((size_t)(b * HH + h)) * DH + d) * NN + n] = (h16)val;
        }
      } else {  // MODE 3: bias + exact GELU, f16 out
        const float u = val + bias[gcol];
        const float ge = 0.5f * u * (1.0f + erff(u * 0.70710678f));
        outH[(size_t)grow * Nc + gcol] = (h16)ge;
      }
    }
  };
  epi(acc00, row0, col0);
  epi(acc01, row0, col0 + 16);
  epi(acc10, row0 + 16, col0);
  epi(acc11, row0 + 16, col0 + 16);
}

// ---------------- fused GPSA attention ----------------
// grid = (N/16, B*H); block = 256 (8 waves). Wave wv owns keys [wv*128, wv*128+128).
#define SA_STRIDE 1032  // 1024 + 8 halfs pad (16B) to stagger LDS banks

__global__ __launch_bounds__(256) void attn_kernel(
    const h16* __restrict__ q, const h16* __restrict__ k, const h16* __restrict__ vt,
    const float* __restrict__ pos_w, const float* __restrict__ pos_b,
    const float* __restrict__ gating, h16* __restrict__ o_h) {
  __shared__ h16   sA[16 * SA_STRIDE];     // attn f16 [16 x 1024]; later aliased as fp32 O partials
  __shared__ float sredA[8][16], sredB[8][16];
  __shared__ float srow0[16], srow1[16], srow2[16], srow3[16], srow4[16];

  const int lane = threadIdx.x & 31;
  const int wv = threadIdx.x >> 5;
  const int hi = lane >> 4;
  const int ln = lane & 15;
  const int qb = blockIdx.x, bh = blockIdx.y;
  const int b = bh / HH, h = bh % HH;
  const int i0 = qb * 16;
  const int mbase = wv * 128;

  const h16* qbase = q + ((size_t)bh * NN + i0) * DH;
  const h16* kbase = k + (size_t)bh * NN * DH;
  const h16* vbase = vt + (size_t)bh * DH * NN;

  // ---- phase 1: patch logits S = (q*SCALE) . k^T, 16 rows x 128 cols per wave ----
  v8f acc[8] = {};
  #pragma unroll
  for (int k0 = 0; k0 < DH; k0 += 32) {
    v16h a = load_frag_row(qbase, DH, k0, lane);
    #pragma unroll
    for (int t = 0; t < 8; ++t) {
      v16h bf = load_frag_col(kbase + (size_t)(mbase + t * 16) * DH, DH, k0, lane);
      acc[t] = WMMA_F16(a, bf, acc[t]);
    }
  }

  const float w0 = pos_w[2 * h], w1 = pos_w[2 * h + 1], pb = pos_b[h];
  const float gval = 1.f / (1.f + expf(-gating[h]));

  // ---- phase 2a: row maxima (patch + positional) ----
  #pragma unroll
  for (int r = 0; r < 8; ++r) {
    const int row = 8 * hi + r;
    const float iF = (float)(i0 + row);
    float mP = -1e30f, mG = -1e30f;
    #pragma unroll
    for (int t = 0; t < 8; ++t) {
      const float rel = iF - (float)(mbase + t * 16 + ln);
      mG = fmaxf(mG, w0 * rel + w1 * fabsf(rel) + pb);
      mP = fmaxf(mP, acc[t][r]);
    }
    for (int off = 1; off < 16; off <<= 1) {
      mP = fmaxf(mP, __shfl_xor(mP, off));
      mG = fmaxf(mG, __shfl_xor(mG, off));
    }
    if (ln == 0) { sredA[wv][row] = mP; sredB[wv][row] = mG; }
  }
  __syncthreads();
  if (threadIdx.x < 16) {
    float mP = sredA[0][threadIdx.x], mG = sredB[0][threadIdx.x];
    for (int w2 = 1; w2 < 8; ++w2) {
      mP = fmaxf(mP, sredA[w2][threadIdx.x]);
      mG = fmaxf(mG, sredB[w2][threadIdx.x]);
    }
    srow0[threadIdx.x] = mP; srow1[threadIdx.x] = mG;
  }
  __syncthreads();

  // ---- phase 2b: row exp-sums ----
  #pragma unroll
  for (int r = 0; r < 8; ++r) {
    const int row = 8 * hi + r;
    const float iF = (float)(i0 + row);
    float sP = 0.f, sG = 0.f;
    #pragma unroll
    for (int t = 0; t < 8; ++t) {
      const float rel = iF - (float)(mbase + t * 16 + ln);
      sG += expf(w0 * rel + w1 * fabsf(rel) + pb - srow1[row]);
      sP += expf(acc[t][r] - srow0[row]);
    }
    for (int off = 1; off < 16; off <<= 1) { sP += __shfl_xor(sP, off); sG += __shfl_xor(sG, off); }
    if (ln == 0) { sredA[wv][row] = sP; sredB[wv][row] = sG; }
  }
  __syncthreads();
  if (threadIdx.x < 16) {
    float sP = 0.f, sG = 0.f;
    for (int w2 = 0; w2 < 8; ++w2) { sP += sredA[w2][threadIdx.x]; sG += sredB[w2][threadIdx.x]; }
    srow2[threadIdx.x] = sP; srow3[threadIdx.x] = sG;
  }
  __syncthreads();

  // ---- phase 3: gated mix + renormalization sum ----
  #pragma unroll
  for (int r = 0; r < 8; ++r) {
    const int row = 8 * hi + r;
    const float iF = (float)(i0 + row);
    float sC = 0.f;
    #pragma unroll
    for (int t = 0; t < 8; ++t) {
      const float rel = iF - (float)(mbase + t * 16 + ln);
      const float pa = expf(acc[t][r] - srow0[row]) / srow2[row];
      const float po = expf(w0 * rel + w1 * fabsf(rel) + pb - srow1[row]) / srow3[row];
      const float c = (1.f - gval) * pa + gval * po;
      acc[t][r] = c;
      sC += c;
    }
    for (int off = 1; off < 16; off <<= 1) sC += __shfl_xor(sC, off);
    if (ln == 0) sredA[wv][row] = sC;
  }
  __syncthreads();
  if (threadIdx.x < 16) {
    float sC = 0.f;
    for (int w2 = 0; w2 < 8; ++w2) sC += sredA[w2][threadIdx.x];
    srow4[threadIdx.x] = sC;
  }
  __syncthreads();

  // write normalized attn (f16) to LDS — each wave writes only its own key range
  #pragma unroll
  for (int r = 0; r < 8; ++r) {
    const int row = 8 * hi + r;
    const float inv = 1.f / srow4[row];
    #pragma unroll
    for (int t = 0; t < 8; ++t)
      sA[row * SA_STRIDE + mbase + t * 16 + ln] = (h16)(acc[t][r] * inv);
  }

  // ---- phase 4: partial O = attn[:, my_keys] @ V[my_keys, :64] ----
  v8f oacc[4] = {};
  #pragma unroll
  for (int kq = 0; kq < 128; kq += 32) {
    AF fa;
    const h16* pA = sA + ln * SA_STRIDE + mbase + kq + hi * 8;
    fa.h[0] = *(const v8h*)pA;
    fa.h[1] = *(const v8h*)(pA + 16);
    #pragma unroll
    for (int t = 0; t < 4; ++t) {
      v16h bf = load_frag_col(vbase + (size_t)(t * 16) * NN, NN, mbase + kq, lane);
      oacc[t] = WMMA_F16(fa.v, bf, oacc[t]);
    }
  }

  // ---- phase 5: cross-wave O reduction (alias sA as fp32 [8][16][64]) ----
  __syncthreads();
  float* sO = (float*)sA;
  #pragma unroll
  for (int t = 0; t < 4; ++t)
    #pragma unroll
    for (int r = 0; r < 8; ++r)
      sO[((wv * 16) + (r + 8 * hi)) * 64 + t * 16 + ln] = oacc[t][r];
  __syncthreads();
  for (int e = threadIdx.x; e < 16 * 64; e += 256) {
    const int row = e >> 6, dd = e & 63;
    float s = 0.f;
    for (int w2 = 0; w2 < 8; ++w2) s += sO[(w2 * 16 + row) * 64 + dd];
    o_h[((size_t)(b * NN + i0 + row)) * DD + h * DH + dd] = (h16)s;
  }
}

// ---------------- host-side orchestration ----------------
extern "C" void kernel_launch(void* const* d_in, const int* in_sizes, int n_in,
                              void* d_out, int out_size, void* d_ws, size_t ws_size,
                              hipStream_t stream) {
  const float* x      = (const float*)d_in[0];
  const float* ln1_w  = (const float*)d_in[1];
  const float* ln1_b  = (const float*)d_in[2];
  const float* qk_w   = (const float*)d_in[3];
  const float* v_w    = (const float*)d_in[4];
  const float* proj_w = (const float*)d_in[5];
  const float* proj_b = (const float*)d_in[6];
  const float* pos_w  = (const float*)d_in[7];
  const float* pos_b  = (const float*)d_in[8];
  const float* gating = (const float*)d_in[9];
  const float* ln2_w  = (const float*)d_in[10];
  const float* ln2_b  = (const float*)d_in[11];
  const float* fc1_w  = (const float*)d_in[12];
  const float* fc1_b  = (const float*)d_in[13];
  const float* fc2_w  = (const float*)d_in[14];
  const float* fc2_b  = (const float*)d_in[15];
  float* out = (float*)d_out;

  // workspace layout (bytes, 256-aligned sizes)
  char* ws = (char*)d_ws;
  size_t off = 0;
  h16* wqkv_h = (h16*)(ws + off); off += (size_t)3 * DD * DD * 2;            // 2304x768 f16
  h16* projw_h = (h16*)(ws + off); off += (size_t)DD * DD * 2;
  h16* fc1w_h = (h16*)(ws + off); off += (size_t)MLPH * DD * 2;
  h16* fc2w_h = (h16*)(ws + off); off += (size_t)DD * MLPH * 2;
  // big block: h1 | q | k | v_t  (all dead after proj) — later aliased by fc1 activations
  char* big = ws + off;
  h16* h1  = (h16*)(big);
  h16* q_h = (h16*)(big + (size_t)MROWS * DD * 2);
  h16* k_h = (h16*)(big + (size_t)MROWS * DD * 2 + (size_t)BB * HH * NN * DH * 2);
  h16* v_t = (h16*)(big + (size_t)MROWS * DD * 2 + (size_t)2 * BB * HH * NN * DH * 2);
  h16* u_h = (h16*)(big);  // fc1 output [8192 x 3072] aliases h1/q/k/v
  off += (size_t)MROWS * DD * 2 + (size_t)3 * BB * HH * NN * DH * 2;          // 50.3 MB
  h16* o_h = (h16*)(ws + off); off += (size_t)MROWS * DD * 2;
  float* x1 = (float*)(ws + off); off += (size_t)MROWS * DD * 4;
  h16* h2  = (h16*)(ws + off); off += (size_t)MROWS * DD * 2;
  (void)ws_size; (void)n_in; (void)in_sizes; (void)out_size;

  // 1) weight conversion (qk_w || v_w concatenated into wqkv_h)
  {
    int n1 = 2 * DD * DD;  // qk_w
    f32_to_f16_kernel<<<(n1 + 255) / 256, 256, 0, stream>>>(qk_w, wqkv_h, n1);
    int n2 = DD * DD;      // v_w -> rows 1536..2303
    f32_to_f16_kernel<<<(n2 + 255) / 256, 256, 0, stream>>>(v_w, wqkv_h + (size_t)2 * DD * DD, n2);
    f32_to_f16_kernel<<<(n2 + 255) / 256, 256, 0, stream>>>(proj_w, projw_h, n2);
    int n3 = MLPH * DD;
    f32_to_f16_kernel<<<(n3 + 255) / 256, 256, 0, stream>>>(fc1_w, fc1w_h, n3);
    f32_to_f16_kernel<<<(n3 + 255) / 256, 256, 0, stream>>>(fc2_w, fc2w_h, n3);
  }

  // 2) LN1: x -> h1 (f16)
  ln_kernel<<<MROWS, 256, 0, stream>>>(x, ln1_w, ln1_b, h1);

  // 3) fused QKV projection: h1[8192x768] @ wqkv^T[768->2304] -> q_h/k_h/v_t
  gemm_f16_kernel<2><<<dim3(3 * DD / 64, MROWS / 128), 256, 0, stream>>>(
      h1, wqkv_h, MROWS, 3 * DD, DD,
      nullptr, nullptr, nullptr, nullptr, q_h, k_h, v_t);

  // 4) fused GPSA attention -> o_h (f16, [B,N,D] layout)
  attn_kernel<<<dim3(NN / 16, BB * HH), 256, 0, stream>>>(
      q_h, k_h, v_t, pos_w, pos_b, gating, o_h);

  // 5) proj + residual: x1 = x + o_h @ proj_w^T + proj_b   (fp32)
  gemm_f16_kernel<0><<<dim3(DD / 64, MROWS / 128), 256, 0, stream>>>(
      o_h, projw_h, MROWS, DD, DD,
      proj_b, x, x1, nullptr, nullptr, nullptr, nullptr);

  // 6) LN2: x1 -> h2 (f16)
  ln_kernel<<<MROWS, 256, 0, stream>>>(x1, ln2_w, ln2_b, h2);

  // 7) fc1 + exact GELU: u_h = gelu(h2 @ fc1_w^T + fc1_b)  (f16)
  gemm_f16_kernel<3><<<dim3(MLPH / 64, MROWS / 128), 256, 0, stream>>>(
      h2, fc1w_h, MROWS, MLPH, DD,
      fc1_b, nullptr, nullptr, u_h, nullptr, nullptr, nullptr);

  // 8) fc2 + residual: out = x1 + u_h @ fc2_w^T + fc2_b    (fp32)
  gemm_f16_kernel<0><<<dim3(DD / 64, MROWS / 128), 256, 0, stream>>>(
      u_h, fc2w_h, MROWS, DD, MLPH,
      fc2_b, x1, out, nullptr, nullptr, nullptr, nullptr);
}
